// DgCnnPN_21612275433711
// MI455X (gfx1250) — compile-verified, matrix operand order checked by reading, not run on hardware
//
#include <hip/hip_runtime.h>

typedef __attribute__((ext_vector_type(16))) __bf16 bf16x16;
typedef __attribute__((ext_vector_type(8)))  __bf16 bf16x8;
typedef __attribute__((ext_vector_type(8)))  float  f32x8;

union FragU { bf16x16 v; bf16x8 h[2]; };

__device__ __forceinline__ float sigmoidf_(float x) {
  return 1.0f / (1.0f + __expf(-x));
}

// ---------------------------------------------------------------------------
// Generic WMMA GEMM:  C[z][m,n] = act( bias[n] + sum_tap sum_k A[z][m+shift,k] * Bt[z][tap][n,k] )
//   A  : bf16 row-major [M, lda], row shift = (tap - taps/2)*dil, valid within Sper period
//   Bt : bf16 [taps][npad, ldb]   (B transposed: Bt[n*ldb + k] = B[k, n])
//   C  : f32  [M, ldc]
//   zeros: >= (maxK+32) bf16 zeros; invalid/padded lanes load from here (no divergence)
//   z  : batched via blockIdx.z with (outer,inner) strides, zinner = inner count
// Each block: 128 threads = 4 waves; wave w -> 16-row tile, block -> NT*16 cols.
// NT is a compile-time constant: fixed accumulator registers, no phi shuffles.
// Inner loop is branch-free: loads + v_wmma only (EXEC all-1s throughout).
// ---------------------------------------------------------------------------
template <int NT>
__global__ __launch_bounds__(128)
void wmma_gemm_t(const __bf16* __restrict__ A, int lda, long sAo, long sAi,
                 const __bf16* __restrict__ Bt, int ldb, long sBo, long sBi, int npad,
                 const float* __restrict__ bias,
                 float* __restrict__ C, int ldc, long sCo, long sCi,
                 const __bf16* __restrict__ zeros,
                 int M, int N, int K, int Sper, int taps, int dil, int act, int zinner)
{
  const int lane = threadIdx.x & 31;
  const int wv   = threadIdx.x >> 5;
  const int m0   = (blockIdx.y * 4 + wv) * 16;
  if (m0 >= M) return;
  const int n0 = blockIdx.x * (NT * 16);

  const long zo = blockIdx.z / zinner;
  const long zi = blockIdx.z % zinner;
  const __bf16* Az = A + zo * sAo + zi * sAi;
  const __bf16* Bz = Bt + zo * sBo + zi * sBi;
  float* Cz = C + zo * sCo + zi * sCi;

  const int nl  = lane & 15;
  const int kh  = lane >> 4;
  const int row = m0 + nl;
  const __bf16* zrow = zeros + kh * 8;

  f32x8 acc[NT];
#pragma unroll
  for (int i = 0; i < NT; ++i) acc[i] = {};

  for (int tap = 0; tap < taps; ++tap) {
    const int shift = (tap - (taps >> 1)) * dil;
    const int s = row % Sper;
    const bool valid = (row < M) && (s + shift >= 0) && (s + shift < Sper);
    // per-lane pointer select instead of divergent zero-fill
    const __bf16* arow = valid ? (Az + (long)(row + shift) * lda + kh * 8) : zrow;
    const __bf16* bbase = Bz + (long)tap * (long)npad * (long)ldb;
    const __bf16* brow[NT];
#pragma unroll
    for (int nt = 0; nt < NT; ++nt) {
      const int n = n0 + nt * 16 + nl;
      brow[nt] = (n < npad) ? (bbase + (long)n * ldb + kh * 8) : zrow;
    }
#pragma unroll 2
    for (int k0 = 0; k0 < K; k0 += 32) {
      FragU a;
      a.h[0] = *(const bf16x8*)(arow + k0);
      a.h[1] = *(const bf16x8*)(arow + k0 + 16);
#pragma unroll
      for (int nt = 0; nt < NT; ++nt) {
        FragU b;
        b.h[0] = *(const bf16x8*)(brow[nt] + k0);
        b.h[1] = *(const bf16x8*)(brow[nt] + k0 + 16);
#if defined(__AMDGCN__)
        acc[nt] = __builtin_amdgcn_wmma_f32_16x16x32_bf16(
            false, a.v, false, b.v, (short)0, acc[nt], false, false);
#endif
      }
    }
  }

#pragma unroll
  for (int nt = 0; nt < NT; ++nt) {
    const int c = n0 + nt * 16 + nl;
    float bv = (bias && c < N) ? bias[c] : 0.0f;
#pragma unroll
    for (int v = 0; v < 8; ++v) {
      const int r = m0 + v + 8 * kh;
      if (r < M && c < N) {
        float val = acc[nt][v] + bv;
        if (act == 1) val = fmaxf(val, 0.0f);
        else if (act == 2) val = sigmoidf_(val);
        Cz[(long)r * ldc + c] = val;
      }
    }
  }
}

// --------------------------- elementwise / pack kernels ---------------------

__global__ void zero_fill_bf16(__bf16* __restrict__ p, long n) {
  long idx = (long)blockIdx.x * blockDim.x + threadIdx.x;
  if (idx >= n) return;
  p[idx] = (__bf16)0.0f;
}

// Bt[n*K + k] = bf16(W[k*N + n]); rows n in [N, npad) zero-padded.
__global__ void pack_wT(const float* __restrict__ W, __bf16* __restrict__ Bt,
                        int K, int N, int npad) {
  long idx = (long)blockIdx.x * blockDim.x + threadIdx.x;
  long total = (long)npad * K;
  if (idx >= total) return;
  int n = (int)(idx / K), k = (int)(idx % K);
  float v = (n < N) ? W[(long)k * N + n] : 0.0f;
  Bt[idx] = (__bf16)v;
}

__global__ void embed_kernel(const int* __restrict__ chars, const int* __restrict__ words,
                             const int* __restrict__ pos,
                             const float* __restrict__ char_emb, const float* __restrict__ word_emb,
                             const float* __restrict__ pos_emb,
                             float* __restrict__ X, float* __restrict__ maskf) {
  int m = blockIdx.x;           // 8192 tokens
  int t = threadIdx.x;          // 256 features
  int c = chars[m], w = words[m], p = pos[m];
  float pe = pos_emb[(long)p * 256 + t];
  float e = (t < 128) ? char_emb[(long)c * 128 + t] : word_emb[(long)w * 128 + (t - 128)];
  X[(long)m * 256 + t] = e + pe;
  if (t == 0) maskf[m] = (c != 0) ? 1.0f : 0.0f;
}

__global__ void to_bf16_masked(const float* __restrict__ X, const float* __restrict__ maskf,
                               __bf16* __restrict__ D, int Cw, long M) {
  long idx = (long)blockIdx.x * blockDim.x + threadIdx.x;
  if (idx >= M * Cw) return;
  long m = idx / Cw;
  D[idx] = (__bf16)(X[idx] * maskf[m]);
}

__global__ void to_bf16(const float* __restrict__ S, __bf16* __restrict__ D, long n) {
  long idx = (long)blockIdx.x * blockDim.x + threadIdx.x;
  if (idx >= n) return;
  D[idx] = (__bf16)S[idx];
}

__global__ void dgc_gate(const float* __restrict__ H, const float* __restrict__ maskf,
                         float* __restrict__ X) {
  long idx = (long)blockIdx.x * blockDim.x + threadIdx.x;
  if (idx >= 8192L * 256) return;
  long m = idx >> 8; int j = (int)(idx & 255);
  float mk = maskf[m];
  const float* h = H + m * 512;
  float g  = sigmoidf_(h[j]);
  float hh = h[256 + j];
  float xm = X[idx] * mk;
  X[idx] = (xm * (1.0f - g) + hh * g) * mk;
}

// QF/KF/VF [8192, H*Kh] f32 -> QH/KH [b,h,q,32] bf16 (pad k>=Kh), VT [b,h,32,512] bf16
__global__ void pack_heads(const float* __restrict__ QF, const float* __restrict__ KF,
                           const float* __restrict__ VF,
                           __bf16* __restrict__ QH, __bf16* __restrict__ KH,
                           __bf16* __restrict__ VT, int Kh, int Hh, long total) {
  long idx = (long)blockIdx.x * blockDim.x + threadIdx.x;
  if (idx >= total) return;
  int k = (int)(idx & 31);
  long r = idx >> 5;           // bh*512 + q
  int q = (int)(r & 511);
  long bh = r >> 9;
  int h = (int)(bh % Hh); long b = bh / Hh;
  long m = b * 512 + q;
  int D = Hh * Kh;
  float qv = 0.f, kv = 0.f, vv = 0.f;
  if (k < Kh) { long o = m * D + (long)h * Kh + k; qv = QF[o]; kv = KF[o]; vv = VF[o]; }
  QH[idx] = (__bf16)qv;
  KH[idx] = (__bf16)kv;
  VT[(bh * 32 + k) * 512 + q] = (__bf16)vv;
}

// one block per row of logits (chunk = 8 heads x 512 q); mask over key positions
__global__ void softmax_mask(const float* __restrict__ SC, const float* __restrict__ maskb,
                             __bf16* __restrict__ PR, float scale) {
  __shared__ float red[256];
  long row = blockIdx.x;
  const float* src = SC + row * 512;
  __bf16* dst = PR + row * 512;
  int t = threadIdx.x;
  float v0 = (maskb[t]       != 0.f) ? src[t]       * scale : -1e9f;
  float v1 = (maskb[t + 256] != 0.f) ? src[t + 256] * scale : -1e9f;
  red[t] = fmaxf(v0, v1); __syncthreads();
  for (int o = 128; o > 0; o >>= 1) { if (t < o) red[t] = fmaxf(red[t], red[t + o]); __syncthreads(); }
  float mx = red[0]; __syncthreads();
  float e0 = __expf(v0 - mx), e1 = __expf(v1 - mx);
  red[t] = e0 + e1; __syncthreads();
  for (int o = 128; o > 0; o >>= 1) { if (t < o) red[t] += red[t + o]; __syncthreads(); }
  float inv = 1.0f / red[0];
  dst[t]       = (__bf16)(e0 * inv);
  dst[t + 256] = (__bf16)(e1 * inv);
}

// concat two f32 matrices along columns into one bf16 matrix
__global__ void concat_bf16(const float* __restrict__ A, int wa,
                            const float* __restrict__ Bv, int wb,
                            __bf16* __restrict__ D, long M) {
  long idx = (long)blockIdx.x * blockDim.x + threadIdx.x;
  int w = wa + wb;
  if (idx >= M * w) return;
  long m = idx / w; int j = (int)(idx % w);
  float v = (j < wa) ? A[m * wa + j] : Bv[m * wb + (j - wa)];
  D[idx] = (__bf16)v;
}

__global__ void sub_out(const float* __restrict__ AD, const float* __restrict__ sub_w,
                        const float* __restrict__ sub_b, float* __restrict__ out) {
  long idx = (long)blockIdx.x * blockDim.x + threadIdx.x;
  if (idx >= 8192L * 2) return;
  int c = (int)(idx & 1); long m = idx >> 1;
  float accv = sub_b[c];
  const float* row = AD + m * 128;
  for (int k = 0; k < 128; ++k) accv += row[k] * sub_w[k * 2 + c];
  long b = m >> 9, s = m & 511;
  out[(b * 2 + c) * 512 + s] = sigmoidf_(accv);
}

__global__ void gather_sse(const float* __restrict__ AD, const int* __restrict__ loc,
                           float* __restrict__ SSE) {
  int b = blockIdx.x;          // 16
  int t = threadIdx.x;         // 256
  int which = t >> 7, j = t & 127;
  int s = loc[b * 2 + which];
  SSE[(long)b * 256 + t] = AD[((long)b * 512 + s) * 128 + j];
}

__global__ void build_pofeat(const float* __restrict__ AD, const float* __restrict__ SSE,
                             float* __restrict__ PF, __bf16* __restrict__ PFBF) {
  long idx = (long)blockIdx.x * blockDim.x + threadIdx.x;
  if (idx >= 8192L * 384) return;
  long m = idx / 384; int j = (int)(idx % 384);
  long b = m >> 9;
  float v = (j < 128) ? AD[m * 128 + j] : SSE[b * 256 + (j - 128)];
  PF[idx] = v;
  PFBF[idx] = (__bf16)v;
}

__global__ void po_out_tr(const float* __restrict__ PO, float* __restrict__ out) {
  long idx = (long)blockIdx.x * blockDim.x + threadIdx.x;
  if (idx >= 8192L * 100) return;
  long m = idx / 100; int c = (int)(idx % 100);
  long b = m >> 9, s = m & 511;
  out[(b * 100 + c) * 512 + s] = PO[idx];
}

__global__ void mask_out_k(const float* __restrict__ maskf, float* __restrict__ out) {
  long idx = (long)blockIdx.x * blockDim.x + threadIdx.x;
  if (idx >= 8192) return;
  out[idx] = maskf[idx];
}

// ------------------------------- host side ----------------------------------

static void launch_gemm(hipStream_t st,
                        const __bf16* A, int lda, long sAo, long sAi,
                        const __bf16* Bt, int ldb, long sBo, long sBi, int npad,
                        const float* bias, float* C, int ldc, long sCo, long sCi,
                        const __bf16* zeros,
                        int M, int N, int K, int Sper, int taps, int dil, int act,
                        int zcount, int zinner) {
  const int n16 = (N + 15) / 16;
  const dim3 blk(128, 1, 1);
  if (n16 == 1) {
    dim3 grid(1, (M + 63) / 64, zcount);
    wmma_gemm_t<1><<<grid, blk, 0, st>>>(A, lda, sAo, sAi, Bt, ldb, sBo, sBi, npad,
                                         bias, C, ldc, sCo, sCi, zeros,
                                         M, N, K, Sper, taps, dil, act, zinner);
  } else if (n16 == 2) {
    dim3 grid(1, (M + 63) / 64, zcount);
    wmma_gemm_t<2><<<grid, blk, 0, st>>>(A, lda, sAo, sAi, Bt, ldb, sBo, sBi, npad,
                                         bias, C, ldc, sCo, sCi, zeros,
                                         M, N, K, Sper, taps, dil, act, zinner);
  } else {
    dim3 grid((n16 + 3) / 4, (M + 63) / 64, zcount);
    wmma_gemm_t<4><<<grid, blk, 0, st>>>(A, lda, sAo, sAi, Bt, ldb, sBo, sBi, npad,
                                         bias, C, ldc, sCo, sCi, zeros,
                                         M, N, K, Sper, taps, dil, act, zinner);
  }
}

static inline dim3 g1(long n) { return dim3((unsigned)((n + 255) / 256), 1, 1); }

extern "C" void kernel_launch(void* const* d_in, const int* in_sizes, int n_in,
                              void* d_out, int out_size, void* d_ws, size_t ws_size,
                              hipStream_t stream) {
  (void)in_sizes; (void)n_in; (void)out_size; (void)ws_size;
  // ---- inputs, JAX pytree alphabetical order ----
  const int*   sub_loc   = (const int*)d_in[0];
  const int*   inp_char  = (const int*)d_in[1];
  const int*   inp_pos   = (const int*)d_in[2];
  const int*   inp_word  = (const int*)d_in[3];
  const float* a1_Wk = (const float*)d_in[4];
  const float* a1_Wo = (const float*)d_in[5];
  const float* a1_Wq = (const float*)d_in[6];
  const float* a1_Wv = (const float*)d_in[7];
  const float* a1_bk = (const float*)d_in[8];
  const float* a1_bo = (const float*)d_in[9];
  const float* a1_bq = (const float*)d_in[10];
  const float* a1_bv = (const float*)d_in[11];
  const float* a2_Wk = (const float*)d_in[12];
  const float* a2_Wo = (const float*)d_in[13];
  const float* a2_Wq = (const float*)d_in[14];
  const float* a2_Wv = (const float*)d_in[15];
  const float* a2_bk = (const float*)d_in[16];
  const float* a2_bo = (const float*)d_in[17];
  const float* a2_bq = (const float*)d_in[18];
  const float* a2_bv = (const float*)d_in[19];
  const float* char_emb = (const float*)d_in[20];
  const float* conv1_b  = (const float*)d_in[21];
  const float* conv1_k  = (const float*)d_in[22];
  const float* conv2_b  = (const float*)d_in[23];
  const float* conv2_k  = (const float*)d_in[24];
  const float* dgc_b[12]; const float* dgc_k[12];
  for (int i = 0; i < 12; ++i) { dgc_b[i] = (const float*)d_in[25 + i]; dgc_k[i] = (const float*)d_in[37 + i]; }
  const float* po_b    = (const float*)d_in[49];
  const float* po_w    = (const float*)d_in[50];
  const float* pos_emb = (const float*)d_in[51];
  const float* sub_b   = (const float*)d_in[52];
  const float* sub_w   = (const float*)d_in[53];
  const float* word_emb = (const float*)d_in[54];

  // ---- workspace carve-out ----
  char* base = (char*)d_ws; size_t off = 0;
  auto alloc = [&](size_t bytes) -> void* {
    void* r = base + off; off = (off + bytes + 255) & ~(size_t)255; return r;
  };
  __bf16* ZB   = (__bf16*)alloc(2048L * 2);          // zero scratch (>= maxK+32 elems)
  __bf16* WDGC = (__bf16*)alloc(12L * 3 * 512 * 256 * 2);
  __bf16* WQ1  = (__bf16*)alloc(128L * 256 * 2);
  __bf16* WK1  = (__bf16*)alloc(128L * 256 * 2);
  __bf16* WV1  = (__bf16*)alloc(128L * 256 * 2);
  __bf16* WO1  = (__bf16*)alloc(256L * 128 * 2);
  __bf16* WQ2  = (__bf16*)alloc(192L * 384 * 2);
  __bf16* WK2  = (__bf16*)alloc(192L * 384 * 2);
  __bf16* WV2  = (__bf16*)alloc(192L * 384 * 2);
  __bf16* WO2  = (__bf16*)alloc(384L * 192 * 2);
  __bf16* WC1  = (__bf16*)alloc(3L * 128 * 512 * 2);
  __bf16* WC2  = (__bf16*)alloc(3L * 256 * 768 * 2);
  __bf16* WPO  = (__bf16*)alloc(112L * 256 * 2);
  float*  MASKF = (float*)alloc(8192L * 4);
  float*  X     = (float*)alloc(8192L * 256 * 4);
  __bf16* XBF   = (__bf16*)alloc(8192L * 256 * 2);
  float*  Hbuf  = (float*)alloc(8192L * 512 * 4);   // also reused for PO [8192,100]
  float*  QF    = (float*)alloc(8192L * 192 * 4);
  float*  KF    = (float*)alloc(8192L * 192 * 4);
  float*  VF    = (float*)alloc(8192L * 192 * 4);
  __bf16* QH    = (__bf16*)alloc(16L * 8 * 512 * 32 * 2);
  __bf16* KH    = (__bf16*)alloc(16L * 8 * 512 * 32 * 2);
  __bf16* VT    = (__bf16*)alloc(16L * 8 * 32 * 512 * 2);
  float*  SCC   = (float*)alloc(8L * 512 * 512 * 4);
  __bf16* PRR   = (__bf16*)alloc(8L * 512 * 512 * 2);
  float*  OH    = (float*)alloc(8192L * 192 * 4);
  __bf16* OHBF  = (__bf16*)alloc(8192L * 192 * 2);
  float*  ATT   = (float*)alloc(8192L * 384 * 4);
  __bf16* CATBF = (__bf16*)alloc(8192L * 768 * 2);
  float*  AD    = (float*)alloc(8192L * 256 * 4);
  __bf16* ADBF  = (__bf16*)alloc(8192L * 256 * 2);
  float*  SSE   = (float*)alloc(16L * 256 * 4);
  float*  PF    = (float*)alloc(8192L * 384 * 4);
  __bf16* PFBF  = (__bf16*)alloc(8192L * 384 * 2);
  float*  PO    = Hbuf;

  // ---- 0. zero scratch (ws is poisoned; must be re-zeroed every call) ----
  zero_fill_bf16<<<g1(2048), 256, 0, stream>>>(ZB, 2048);

  // ---- 1. pack all weights into bf16 [N,K] transposed layouts ----
  for (int i = 0; i < 12; ++i)
    for (int t = 0; t < 3; ++t)
      pack_wT<<<g1(512L * 256), 256, 0, stream>>>(dgc_k[i] + (long)t * 256 * 512,
                                                  WDGC + ((long)i * 3 + t) * 512 * 256, 256, 512, 512);
  pack_wT<<<g1(128L * 256), 256, 0, stream>>>(a1_Wq, WQ1, 256, 128, 128);
  pack_wT<<<g1(128L * 256), 256, 0, stream>>>(a1_Wk, WK1, 256, 128, 128);
  pack_wT<<<g1(128L * 256), 256, 0, stream>>>(a1_Wv, WV1, 256, 128, 128);
  pack_wT<<<g1(256L * 128), 256, 0, stream>>>(a1_Wo, WO1, 128, 256, 256);
  pack_wT<<<g1(192L * 384), 256, 0, stream>>>(a2_Wq, WQ2, 384, 192, 192);
  pack_wT<<<g1(192L * 384), 256, 0, stream>>>(a2_Wk, WK2, 384, 192, 192);
  pack_wT<<<g1(192L * 384), 256, 0, stream>>>(a2_Wv, WV2, 384, 192, 192);
  pack_wT<<<g1(384L * 192), 256, 0, stream>>>(a2_Wo, WO2, 192, 384, 384);
  for (int t = 0; t < 3; ++t)
    pack_wT<<<g1(128L * 512), 256, 0, stream>>>(conv1_k + (long)t * 512 * 128,
                                                WC1 + (long)t * 128 * 512, 512, 128, 128);
  for (int t = 0; t < 3; ++t)
    pack_wT<<<g1(256L * 768), 256, 0, stream>>>(conv2_k + (long)t * 768 * 256,
                                                WC2 + (long)t * 256 * 768, 768, 256, 256);
  pack_wT<<<g1(112L * 256), 256, 0, stream>>>(po_w, WPO, 256, 100, 112);

  // ---- 2. embeddings + mask ----
  embed_kernel<<<8192, 256, 0, stream>>>(inp_char, inp_word, inp_pos,
                                         char_emb, word_emb, pos_emb, X, MASKF);

  // ---- 3. 12 dilated gated conv layers ----
  static const int DIL[12] = {1, 2, 5, 1, 2, 5, 1, 2, 5, 1, 1, 1};
  for (int i = 0; i < 12; ++i) {
    to_bf16_masked<<<g1(8192L * 256), 256, 0, stream>>>(X, MASKF, XBF, 256, 8192);
    launch_gemm(stream, XBF, 256, 0, 0, WDGC + (long)i * 3 * 512 * 256, 256, 0, 0, 512,
                dgc_b[i], Hbuf, 512, 0, 0, ZB, 8192, 512, 256, 512, 3, DIL[i], 0, 1, 1);
    dgc_gate<<<g1(8192L * 256), 256, 0, stream>>>(Hbuf, MASKF, X);
  }
  to_bf16_masked<<<g1(8192L * 256), 256, 0, stream>>>(X, MASKF, XBF, 256, 8192);

  // ---- 4. MHA #1 (H=8, K=16 padded to 32) ----
  launch_gemm(stream, XBF, 256, 0, 0, WQ1, 256, 0, 0, 128, a1_bq, QF, 128, 0, 0,
              ZB, 8192, 128, 256, 512, 1, 0, 0, 1, 1);
  launch_gemm(stream, XBF, 256, 0, 0, WK1, 256, 0, 0, 128, a1_bk, KF, 128, 0, 0,
              ZB, 8192, 128, 256, 512, 1, 0, 0, 1, 1);
  launch_gemm(stream, XBF, 256, 0, 0, WV1, 256, 0, 0, 128, a1_bv, VF, 128, 0, 0,
              ZB, 8192, 128, 256, 512, 1, 0, 0, 1, 1);
  pack_heads<<<g1(16L * 8 * 512 * 32), 256, 0, stream>>>(QF, KF, VF, QH, KH, VT, 16, 8,
                                                         16L * 8 * 512 * 32);
  for (int b = 0; b < 16; ++b) {
    const long hs = 512L * 32;
    launch_gemm(stream, QH + (long)b * 8 * hs, 32, 0, hs,
                KH + (long)b * 8 * hs, 32, 0, hs, 512,
                nullptr, SCC, 512, 0, 512L * 512,
                ZB, 512, 512, 32, 512, 1, 0, 0, 8, 8);
    softmax_mask<<<8 * 512, 256, 0, stream>>>(SCC, MASKF + (long)b * 512, PRR, 0.25f);
    launch_gemm(stream, PRR, 512, 0, 512L * 512,
                VT + (long)b * 8 * 32 * 512, 512, 0, 32L * 512, 32,
                nullptr, OH + (long)b * 512 * 128, 128, 0, 16,
                ZB, 512, 16, 512, 512, 1, 0, 0, 8, 8);
  }
  to_bf16<<<g1(8192L * 128), 256, 0, stream>>>(OH, OHBF, 8192L * 128);
  launch_gemm(stream, OHBF, 128, 0, 0, WO1, 128, 0, 0, 256, a1_bo, ATT, 256, 0, 0,
              ZB, 8192, 256, 128, 512, 1, 0, 0, 1, 1);

  // ---- 5. conv1 (relu) + sub head + sse gather + po_feat ----
  concat_bf16<<<g1(8192L * 512), 256, 0, stream>>>(X, 256, ATT, 256, CATBF, 8192);
  launch_gemm(stream, CATBF, 512, 0, 0, WC1, 512, 0, 0, 128, conv1_b, AD, 128, 0, 0,
              ZB, 8192, 128, 512, 512, 3, 1, 1, 1, 1);
  sub_out<<<g1(8192L * 2), 256, 0, stream>>>(AD, sub_w, sub_b, (float*)d_out);
  gather_sse<<<16, 256, 0, stream>>>(AD, sub_loc, SSE);
  build_pofeat<<<g1(8192L * 384), 256, 0, stream>>>(AD, SSE, PF, PFBF);

  // ---- 6. MHA #2 (H=8, K=24 padded to 32) ----
  launch_gemm(stream, PFBF, 384, 0, 0, WQ2, 384, 0, 0, 192, a2_bq, QF, 192, 0, 0,
              ZB, 8192, 192, 384, 512, 1, 0, 0, 1, 1);
  launch_gemm(stream, PFBF, 384, 0, 0, WK2, 384, 0, 0, 192, a2_bk, KF, 192, 0, 0,
              ZB, 8192, 192, 384, 512, 1, 0, 0, 1, 1);
  launch_gemm(stream, PFBF, 384, 0, 0, WV2, 384, 0, 0, 192, a2_bv, VF, 192, 0, 0,
              ZB, 8192, 192, 384, 512, 1, 0, 0, 1, 1);
  pack_heads<<<g1(16L * 8 * 512 * 32), 256, 0, stream>>>(QF, KF, VF, QH, KH, VT, 24, 8,
                                                         16L * 8 * 512 * 32);
  for (int b = 0; b < 16; ++b) {
    const long hs = 512L * 32;
    launch_gemm(stream, QH + (long)b * 8 * hs, 32, 0, hs,
                KH + (long)b * 8 * hs, 32, 0, hs, 512,
                nullptr, SCC, 512, 0, 512L * 512,
                ZB, 512, 512, 32, 512, 1, 0, 0, 8, 8);
    softmax_mask<<<8 * 512, 256, 0, stream>>>(SCC, MASKF + (long)b * 512, PRR, 0.2041241452f);
    launch_gemm(stream, PRR, 512, 0, 512L * 512,
                VT + (long)b * 8 * 32 * 512, 512, 0, 32L * 512, 32,
                nullptr, OH + (long)b * 512 * 192, 192, 0, 24,
                ZB, 512, 24, 512, 512, 1, 0, 0, 8, 8);
  }
  to_bf16<<<g1(8192L * 192), 256, 0, stream>>>(OH, OHBF, 8192L * 192);
  launch_gemm(stream, OHBF, 192, 0, 0, WO2, 192, 0, 0, 384, a2_bo, ATT, 384, 0, 0,
              ZB, 8192, 384, 192, 512, 1, 0, 0, 1, 1);

  // ---- 7. conv2 (relu) + po head + mask output ----
  concat_bf16<<<g1(8192L * 768), 256, 0, stream>>>(ATT, 384, PF, 384, CATBF, 8192);
  launch_gemm(stream, CATBF, 768, 0, 0, WC2, 768, 0, 0, 256, conv2_b, AD, 256, 0, 0,
              ZB, 8192, 256, 768, 512, 3, 1, 1, 1, 1);
  to_bf16<<<g1(8192L * 256), 256, 0, stream>>>(AD, ADBF, 8192L * 256);
  launch_gemm(stream, ADBF, 256, 0, 0, WPO, 256, 0, 0, 112, po_b, PO, 100, 0, 0,
              ZB, 8192, 100, 256, 512, 1, 0, 2, 1, 1);
  po_out_tr<<<g1(8192L * 100), 256, 0, stream>>>(PO, (float*)d_out + 16384);
  mask_out_k<<<g1(8192), 256, 0, stream>>>(MASKF, (float*)d_out + 16384 + 819200);
}
